// HighwayLSTM_84971632984186
// MI455X (gfx1250) — matrix operand order repaired and checked by reading, not compiled, not run on hardware
//
#include <hip/hip_runtime.h>
#include <hip/hip_bf16.h>
#include <stddef.h>

// Highway-LSTM for MI455X (gfx1250), wave32 + WMMA bf16.
// B=64, T=256, H=512, 4 layers, insize==H for all layers.

#define HDIM 512
#define TDIM 256
#define BDIM 64
#define NLAYERS 4
#define NG 3072   // 6*H  (W_ih output width)
#define NH 2560   // 5*H  (W_hh output width / gate block)
#define KD 512    // reduction dim (insize == hidden == 512)
#define APAD 48   // LDS row pitch in bf16 elems (32 data + 16 pad; keeps 16B alignment)

// gfx1250 async global->LDS staging (ASYNCcnt path), with compile-time fallback
// to explicit global_load_b128 + ds_store_b128 if the builtins are absent.
#if __has_builtin(__builtin_amdgcn_global_load_async_to_lds_b128) && \
    __has_builtin(__builtin_amdgcn_s_wait_asynccnt)
#define USE_ASYNC_LDS 1
#else
#define USE_ASYNC_LDS 0
#endif

typedef __bf16 bf16_t;
typedef __attribute__((ext_vector_type(16))) __bf16 v16bf;
typedef __attribute__((ext_vector_type(8)))  float        v8f;
typedef __attribute__((ext_vector_type(4)))  unsigned int u32x4;
typedef __attribute__((ext_vector_type(4)))  int          i32x4;

#if USE_ASYNC_LDS
// Builtin expects: (AS1 int4*, AS3 int4*, imm offset, imm cpol)
typedef __attribute__((address_space(1))) i32x4* gv4_t;
typedef __attribute__((address_space(3))) i32x4* lv4_t;
#endif

__device__ __forceinline__ bf16_t f2bf(float f) {
  union { float f; unsigned u; } in; in.f = f;
  unsigned u = in.u;
  u += 0x7FFFu + ((u >> 16) & 1u);          // round-to-nearest-even
  union { unsigned short s; bf16_t b; } out; out.s = (unsigned short)(u >> 16);
  return out.b;
}

__device__ __forceinline__ float sigmoidf_(float x) {
  return 1.0f / (1.0f + __expf(-x));
}

// ---------------- weight convert + transpose: fp32 W[K x N] -> bf16 Wt[N x K] ----------------
__global__ void __launch_bounds__(256)
k_cvt_wT(const float* __restrict__ w, bf16_t* __restrict__ wt, int N) {
  __shared__ float tile[32][33];
  const int n0 = blockIdx.x * 32;
  const int k0 = blockIdx.y * 32;
  #pragma unroll
  for (int r = 0; r < 4; ++r)
    tile[threadIdx.y + r * 8][threadIdx.x] =
        w[((size_t)(k0 + threadIdx.y + r * 8)) * N + n0 + threadIdx.x];
  __syncthreads();
  #pragma unroll
  for (int r = 0; r < 4; ++r)
    wt[((size_t)(n0 + threadIdx.y + r * 8)) * KD + k0 + threadIdx.x] =
        f2bf(tile[threadIdx.x][threadIdx.y + r * 8]);
}

// ---------------- layer-0 input pack: transpose (B,T,H)->(T,B,H), mask, bf16 ----------------
__global__ void k_pack0(const float* __restrict__ in, const int* __restrict__ len,
                        bf16_t* __restrict__ xb) {
  int i  = blockIdx.x * blockDim.x + threadIdx.x;  // over T*B*H
  int d  = i & (HDIM - 1);
  int tb = i >> 9;
  int b  = tb & (BDIM - 1);
  int t  = tb >> 6;
  float v = (t < len[b]) ? in[((size_t)b * TDIM + t) * HDIM + d] : 0.0f;
  xb[i] = f2bf(v);
}

// ---------------- layer>0 pack: gather prev-layer h (with optional padded reversal) ----------------
__global__ void k_pack(const float* __restrict__ h, const int* __restrict__ len,
                       bf16_t* __restrict__ xb, int rev) {
  int i  = blockIdx.x * blockDim.x + threadIdx.x;
  int d  = i & (HDIM - 1);
  int tb = i >> 9;
  int b  = tb & (BDIM - 1);
  int t  = tb >> 6;
  int tt = t;
  if (rev) { int L = len[b]; tt = (t < L) ? (L - 1 - t) : t; }
  xb[i] = f2bf(h[((size_t)tt * BDIM + b) * HDIM + d]);
}

// ---------------- zero h/c state ----------------
__global__ void k_init(float* __restrict__ c, bf16_t* __restrict__ hb) {
  int i = blockIdx.x * blockDim.x + threadIdx.x;
  if (i < BDIM * HDIM) { c[i] = 0.0f; hb[i] = f2bf(0.0f); }
}

// ---------------- templated WMMA GEMM core ----------------
// C[M x N] (+= epilogue) = A[M x 512] @ Wt^T, with Wt stored pre-transposed [N x 512] bf16.
// BM x BN = workgroup tile; WM x WN = 16x16 subtiles per wave (8 waves, 256 threads).
template <int BM, int BN, int WM, int WN, bool EPI>
__global__ void __launch_bounds__(256)
k_gemm(const bf16_t* __restrict__ A, const bf16_t* __restrict__ Wt,
       float* __restrict__ Out, int ldOut,
       const float* __restrict__ XiT, const float* __restrict__ bias) {
  static_assert((BM / (WM * 16)) * (BN / (WN * 16)) == 8, "8 waves");
  __shared__ __align__(16) bf16_t sA[BM * APAD];
  __shared__ __align__(16) bf16_t sB[BN * APAD];
  const int tid  = threadIdx.x;
  const int lane = tid & 31;
  const int wave = tid >> 5;
  constexpr int NWN = BN / (WN * 16);
  const int mw = wave / NWN;
  const int nw = wave % NWN;
  const int m0 = blockIdx.x * BM;
  const int n0 = blockIdx.y * BN;

  v8f acc[WM][WN];
  #pragma unroll
  for (int i = 0; i < WM; ++i)
    #pragma unroll
    for (int j = 0; j < WN; ++j)
      acc[i][j] = (v8f){0.f,0.f,0.f,0.f,0.f,0.f,0.f,0.f};

  for (int k0 = 0; k0 < KD; k0 += 32) {
    __syncthreads();
    // stage A: BM x 32 bf16, one 16B transfer per thread per 64 rows
    #pragma unroll
    for (int r = 0; r < BM / 64; ++r) {
      int idx = tid + r * 256;
      int row = idx >> 2, kc = (idx & 3) * 8;
#if USE_ASYNC_LDS
      __builtin_amdgcn_global_load_async_to_lds_b128(
          (gv4_t)(A + ((size_t)(m0 + row)) * KD + k0 + kc),
          (lv4_t)(sA + row * APAD + kc), 0, 0);
#else
      *(u32x4*)(sA + row * APAD + kc) =
          *(const u32x4*)(A + ((size_t)(m0 + row)) * KD + k0 + kc);
#endif
    }
    // stage B: BN x 32 bf16 from pre-transposed weights (rows are K-contiguous)
    #pragma unroll
    for (int r = 0; r < BN / 64; ++r) {
      int idx = tid + r * 256;
      int row = idx >> 2, kc = (idx & 3) * 8;
#if USE_ASYNC_LDS
      __builtin_amdgcn_global_load_async_to_lds_b128(
          (gv4_t)(Wt + ((size_t)(n0 + row)) * KD + k0 + kc),
          (lv4_t)(sB + row * APAD + kc), 0, 0);
#else
      *(u32x4*)(sB + row * APAD + kc) =
          *(const u32x4*)(Wt + ((size_t)(n0 + row)) * KD + k0 + kc);
#endif
    }
    if (k0 + 32 < KD) {  // gfx1250 global_prefetch of next K tile
      int row = (tid & 255) >> 2, kc = (tid & 3) * 8;
      __builtin_prefetch((const void*)(A  + ((size_t)(m0 + row)) * KD + k0 + 32 + kc), 0, 1);
      __builtin_prefetch((const void*)(Wt + ((size_t)(n0 + row)) * KD + k0 + 32 + kc), 0, 1);
    }
#if USE_ASYNC_LDS
    __builtin_amdgcn_s_wait_asynccnt(0);
#endif
    __syncthreads();

    const int khalf = lane >> 4;
    union F { v16bf v; u32x4 q[2]; };
    F af[WM], bfr[WN];
    #pragma unroll
    for (int i = 0; i < WM; ++i) {
      const int am = (mw * WM + i) * 16 + (lane & 15);
      af[i].q[0] = *(const u32x4*)(sA + am * APAD + khalf * 8);       // K 0-7 / 8-15
      af[i].q[1] = *(const u32x4*)(sA + am * APAD + 16 + khalf * 8);  // K 16-23 / 24-31
    }
    #pragma unroll
    for (int j = 0; j < WN; ++j) {
      const int bn = (nw * WN + j) * 16 + (lane & 15);
      bfr[j].q[0] = *(const u32x4*)(sB + bn * APAD + khalf * 8);       // K 0-7 / 8-15
      bfr[j].q[1] = *(const u32x4*)(sB + bn * APAD + 16 + khalf * 8);  // K 16-23 / 24-31
    }
    #pragma unroll
    for (int i = 0; i < WM; ++i)
      #pragma unroll
      for (int j = 0; j < WN; ++j)
        acc[i][j] = __builtin_amdgcn_wmma_f32_16x16x32_bf16(
            false, af[i].v, false, bfr[j].v, (short)0, acc[i][j], false, false);
  }

  // C layout: VGPR j -> row j (lanes 0-15) / row j+8 (lanes 16-31), col = lane&15
  #pragma unroll
  for (int i = 0; i < WM; ++i) {
    const int rb = m0 + (mw * WM + i) * 16 + ((lane >> 4) << 3);
    #pragma unroll
    for (int j = 0; j < WN; ++j) {
      const int col = n0 + (nw * WN + j) * 16 + (lane & 15);
      #pragma unroll
      for (int e = 0; e < 8; ++e) {
        float v = acc[i][j][e];
        if (EPI) v += XiT[((size_t)(rb + e)) * NG + col] + bias[col];
        Out[((size_t)(rb + e)) * ldOut + col] = v;
      }
    }
  }
}

// ---------------- per-step gates / highway / state update ----------------
__global__ void k_point(const float* __restrict__ G, const float* __restrict__ XiT,
                        float* __restrict__ C, bf16_t* __restrict__ Hb,
                        float* __restrict__ Hout, const int* __restrict__ len,
                        int s, int rev) {
  int i = blockIdx.x * blockDim.x + threadIdx.x;  // over B*H
  int d = i & (HDIM - 1);
  int b = i >> 9;
  size_t gb = (size_t)b * NH;
  float fi = sigmoidf_(G[gb + d]);
  float ff = sigmoidf_(G[gb + HDIM + d]);
  float gc = tanhf    (G[gb + 2 * HDIM + d]);
  float fo = sigmoidf_(G[gb + 3 * HDIM + d]);
  float fr = sigmoidf_(G[gb + 4 * HDIM + d]);
  float lin = XiT[(size_t)b * NG + NH + d];
  int L = len[b];
  float m = (s < L) ? 1.0f : 0.0f;
  float cn = (ff * C[i] + fi * gc) * m;
  float hn = (fr * (fo * tanhf(cn)) + (1.0f - fr) * lin) * m;
  C[i]  = cn;
  Hb[i] = f2bf(hn);
  int ot = rev ? ((s < L) ? (L - 1 - s) : s) : s;
  Hout[((size_t)ot * BDIM + b) * HDIM + d] = hn;
}

// ---------------- final transpose (T,B,H) -> (B,T,H) ----------------
__global__ void k_final(const float* __restrict__ H, float* __restrict__ out) {
  int i  = blockIdx.x * blockDim.x + threadIdx.x;
  int d  = i & (HDIM - 1);
  int tb = i >> 9;
  int b  = tb & (BDIM - 1);
  int t  = tb >> 6;
  out[((size_t)b * TDIM + t) * HDIM + d] = H[((size_t)t * BDIM + b) * HDIM + d];
}

extern "C" void kernel_launch(void* const* d_in, const int* in_sizes, int n_in,
                              void* d_out, int out_size, void* d_ws, size_t ws_size,
                              hipStream_t stream) {
  (void)in_sizes; (void)n_in; (void)out_size; (void)ws_size;
  const float* inp  = (const float*)d_in[0];
  const float* w    = (const float*)d_in[1];
  const float* bias = (const float*)d_in[2];
  const int*   len  = (const int*)d_in[3];
  float* out = (float*)d_out;

  // carve workspace (256B-aligned regions)
  char* base = (char*)d_ws;
  size_t off = 0;
  auto carve = [&](size_t bytes) -> char* {
    char* p = base + off;
    off = (off + bytes + 255) & ~(size_t)255;
    return p;
  };
  const size_t WIH = (size_t)KD * NG;            // 512*3072 per-layer W_ih elems
  const size_t WHH = (size_t)KD * NH;            // 512*2560 per-layer W_hh elems
  const size_t LW  = WIH + WHH;
  bf16_t* wtb = (bf16_t*)carve(NLAYERS * LW * sizeof(bf16_t));               // ~23 MB (transposed)
  bf16_t* xb  = (bf16_t*)carve((size_t)TDIM * BDIM * HDIM * sizeof(bf16_t)); // 16 MB
  bf16_t* hbf = (bf16_t*)carve((size_t)BDIM * HDIM * sizeof(bf16_t));
  float*  xi  = (float*)carve((size_t)TDIM * BDIM * NG * sizeof(float));     // 201 MB
  float*  g   = (float*)carve((size_t)BDIM * NH * sizeof(float));
  float*  c   = (float*)carve((size_t)BDIM * HDIM * sizeof(float));
  float*  hA  = (float*)carve((size_t)TDIM * BDIM * HDIM * sizeof(float));   // 33 MB

  // convert + transpose all weights once: W[K x N] fp32 -> Wt[N x K] bf16
  for (int l = 0; l < NLAYERS; ++l) {
    const float* wih_f = w + (size_t)l * LW;
    const float* whh_f = wih_f + WIH;
    bf16_t* wtih = wtb + (size_t)l * LW;        // [NG x KD]
    bf16_t* wthh = wtih + WIH;                  // [NH x KD]
    k_cvt_wT<<<dim3(NG / 32, KD / 32), dim3(32, 8), 0, stream>>>(wih_f, wtih, NG);
    k_cvt_wT<<<dim3(NH / 32, KD / 32), dim3(32, 8), 0, stream>>>(whh_f, wthh, NH);
  }

  const int nTBH = TDIM * BDIM * HDIM;
  for (int l = 0; l < NLAYERS; ++l) {
    const int rev = l & 1;
    if (l == 0) k_pack0<<<nTBH / 256, 256, 0, stream>>>(inp, len, xb);
    else        k_pack <<<nTBH / 256, 256, 0, stream>>>(hA, len, xb, rev);

    const bf16_t* wtih = wtb + (size_t)l * LW;
    const bf16_t* wthh = wtih + WIH;

    // Xi[16384 x 3072] = Xb @ W_ih  : 128x64 WG tile, 2x2 subtiles/wave
    k_gemm<128, 64, 2, 2, false>
        <<<dim3((TDIM * BDIM) / 128, NG / 64), 256, 0, stream>>>(
            xb, wtih, xi, NG, nullptr, nullptr);

    k_init<<<(BDIM * HDIM) / 256, 256, 0, stream>>>(c, hbf);

    const float* bl = bias + (size_t)l * NH;
    for (int s = 0; s < TDIM; ++s) {
      const float* xiT = xi + (size_t)s * BDIM * NG;
      // G[64 x 2560] = Hb @ W_hh + Xi_t + bias : 64x64 WG tile, 2x1 subtiles/wave
      k_gemm<64, 64, 2, 1, true>
          <<<dim3(1, NH / 64), 256, 0, stream>>>(hbf, wthh, g, NH, xiT, bl);
      k_point<<<(BDIM * HDIM) / 256, 256, 0, stream>>>(g, xiT, c, hbf, hA, len, s, rev);
    }
  }
  k_final<<<nTBH / 256, 256, 0, stream>>>(hA, out);
}
